// FixedDGCNNBackbone_70660801954485
// MI455X (gfx1250) — matrix operand order repaired and checked by reading, not compiled, not run on hardware
//
#include <hip/hip_runtime.h>
#include <hip/hip_bf16.h>

typedef __attribute__((ext_vector_type(16))) _Float16 f16x16;
typedef __attribute__((ext_vector_type(8)))  _Float16 f16x8;
typedef __attribute__((ext_vector_type(8)))  float    f32x8;

#define EPS_BN 1e-5f
#define SLOPE  0.2f

static constexpr int BB = 8;      // batch
static constexpr int NP = 4096;   // points
static constexpr int KN = 20;     // knn k

// ---------------------------------------------------------------------------
// Fragment layout used here (wave32 WMMA 16x16x32 f16; A/B in contiguous k, not AMD's table order):
//   A (16x32, MxK): element h of lane l  ->  M = l%16, K = 16*(l/16) + h
//   B (32x16, KxN): element h of lane l  ->  K = 16*(l/16) + h, N = l%16
//   C/D (16x16 f32): VGPR r of lane l    ->  M = r + 8*(l/16),  N = l%16
// ---------------------------------------------------------------------------

// (B,9,N) f32 -> (B,N,16) f32, channels 9..15 zero
__global__ __launch_bounds__(256) void transpose_pad_kernel(const float* __restrict__ x,
                                                            float* __restrict__ xt) {
  int i = blockIdx.x * 256 + threadIdx.x;            // over B*N*16
  if (i >= BB * NP * 16) return;
  int c = i & 15;
  int n = (i >> 4) % NP;
  int b = (i >> 4) / NP;
  xt[i] = (c < 9) ? x[((size_t)b * 9 + c) * NP + n] : 0.f;
}

// Pack row-major W (Cout x Kreal) f32 into B-fragment-ordered f16:
// frag (kt,nt) at ((kt*ntiles)+nt)*512, element e=lane*16+h.
__global__ __launch_bounds__(256) void pack_b_kernel(const float* __restrict__ W,
                                                     _Float16* __restrict__ dst,
                                                     int Cout, int Kreal, int Kpad, int Npad) {
  int ntiles = Npad >> 4;
  int total = (Kpad >> 5) * ntiles * 512;
  int i = blockIdx.x * 256 + threadIdx.x;
  if (i >= total) return;
  int e = i & 511, frag = i >> 9;
  int nt = frag % ntiles, kt = frag / ntiles;
  int lane = e >> 4, h = e & 15;
  int K = kt * 32 + 16 * (lane >> 4) + h;
  int N = nt * 16 + (lane & 15);
  float v = (K < Kreal && N < Cout) ? W[(size_t)N * Kreal + K] : 0.f;
  dst[i] = (_Float16)v;
}

// Fold BN (+optional conv bias) into per-column scale/offset.
__global__ __launch_bounds__(256) void bn_params_kernel(const float* __restrict__ g,
                                                        const float* __restrict__ beta,
                                                        const float* __restrict__ bias,
                                                        float* __restrict__ sc,
                                                        float* __restrict__ of,
                                                        int n, int nreal) {
  int j = blockIdx.x * 256 + threadIdx.x;
  if (j >= n) return;
  if (j >= nreal) { sc[j] = 1.f; of[j] = 0.f; return; }
  float s = g ? g[j] * rsqrtf(1.f + EPS_BN) : 1.f;
  sc[j] = s;
  of[j] = (bias ? bias[j] * s : 0.f) + (beta ? beta[j] : 0.f);
}

// ---------------------------------------------------------------------------
// kNN: one thread per query point, candidates tiled through LDS, top-20 kept
// entirely in registers (unrolled conditional updates -> no scratch).
// ---------------------------------------------------------------------------
template <int C>
__global__ __launch_bounds__(128) void knn_kernel(const float* __restrict__ feat,
                                                  int* __restrict__ idx) {
  extern __shared__ char smemraw[];
  float* tile = (float*)smemraw;                     // TILE * C floats
  constexpr int TILE = 64;
  int b = blockIdx.y;
  int n = blockIdx.x * 128 + threadIdx.x;
  const float* base = feat + (size_t)b * NP * C;
  float q[C];
#pragma unroll
  for (int c = 0; c < C; ++c) q[c] = base[(size_t)n * C + c];

  float d[KN]; int id[KN];
#pragma unroll
  for (int i = 0; i < KN; ++i) { d[i] = 3.4e38f; id[i] = 0; }
  float dmax = 3.4e38f; int mpos = 0;

  for (int t0 = 0; t0 < NP; t0 += TILE) {
    __syncthreads();
    for (int i = threadIdx.x; i < TILE * C; i += 128)
      tile[i] = base[(size_t)(t0 + i / C) * C + (i % C)];
    __syncthreads();
    for (int j = 0; j < TILE; ++j) {
      float dist = 0.f;
#pragma unroll
      for (int c = 0; c < C; ++c) {
        float t = q[c] - tile[j * C + c];
        dist = fmaf(t, t, dist);
      }
      if (dist < dmax) {
#pragma unroll
        for (int i = 0; i < KN; ++i) if (i == mpos) { d[i] = dist; id[i] = t0 + j; }
        dmax = d[0]; mpos = 0;
#pragma unroll
        for (int i = 1; i < KN; ++i) if (d[i] > dmax) { dmax = d[i]; mpos = i; }
      }
    }
  }
  int* o = idx + ((size_t)b * NP + n) * KN;
#pragma unroll
  for (int i = 0; i < KN; ++i) o[i] = id[i];
}

// ---------------------------------------------------------------------------
// EdgeConv: per workgroup, P=8 points. Build E (32 k-rows x KPAD edge feats,
// rows >=20 replicate neighbor 19) in LDS in A-fragment order, then WMMA
// against fragment-ordered weights, fused BN+LeakyReLU+max-over-k epilogue.
// Work loop has a compile-time trip count and scalarized wave id so WMMA is
// never executed under a vector-predicated EXEC mask.
// ---------------------------------------------------------------------------
template <int CIN, int STRIDE, int COUT>
__global__ __launch_bounds__(256) void edgeconv_kernel(const float* __restrict__ feat,
                                                       const int* __restrict__ idx,
                                                       const _Float16* __restrict__ Wfrag,
                                                       const float* __restrict__ sc,
                                                       const float* __restrict__ of,
                                                       float* __restrict__ outF,
                                                       _Float16* __restrict__ cat,
                                                       int catOff) {
  constexpr int KPAD = ((2 * CIN + 31) / 32) * 32;
  constexpr int KT = KPAD / 32;
  constexpr int NT = COUT / 16;
  constexpr int P = 8;
  constexpr int ITERS = (P * NT) / 8;                // uniform per-wave work count
  extern __shared__ char smemraw[];
  _Float16* sE = (_Float16*)smemraw;                 // P * 2 * KT * 512 halves

  const int tid = threadIdx.x;
  const int wv = __builtin_amdgcn_readfirstlane(tid >> 5);   // wave id (SGPR)
  const int ln = tid & 31;                                   // lane id

  // ---- build phase: thread (p=tid/32, k=tid%32) writes one 32-wide row group ----
  {
    const int p = tid >> 5, k = tid & 31;
    const long g = (long)blockIdx.x * P + p;
    const long b = g / NP;
    const int kk = (k < KN) ? k : (KN - 1);
    const int j = idx[g * KN + kk];
    const float* ctr = feat + g * STRIDE;
    const float* nbr = feat + (b * NP + j) * STRIDE;
    const int mt = k >> 4;
    for (int c8 = 0; c8 < KPAD / 8; ++c8) {
      const int kt = c8 >> 2, cc = c8 & 3;
      f16x8 vals;
#pragma unroll
      for (int i2 = 0; i2 < 8; ++i2) {
        int c = c8 * 8 + i2;
        float v;
        if (c < CIN)          v = nbr[c] - ctr[c];
        else if (c < 2 * CIN) v = ctr[c - CIN];
        else                  v = 0.f;
        vals[i2] = (_Float16)v;
      }
      const int lane = (k & 15) + 16 * (cc >> 1);
      const int hb = 8 * (cc & 1);
      *(f16x8*)(sE + (size_t)((p * 2 + mt) * KT + kt) * 512 + lane * 16 + hb) = vals;
    }
  }
  __syncthreads();

  // ---- compute phase: fixed trip count, wave-uniform (pp, nt) ----
  for (int q = 0; q < ITERS; ++q) {
    const int it = q * 8 + wv;
    const int pp = it / NT, nt = it % NT;
    f32x8 a0 = {}; f32x8 a1 = {};
    for (int kt = 0; kt < KT; ++kt) {
      f16x16 af0 = *(const f16x16*)(sE + (size_t)((pp * 2 + 0) * KT + kt) * 512 + ln * 16);
      f16x16 af1 = *(const f16x16*)(sE + (size_t)((pp * 2 + 1) * KT + kt) * 512 + ln * 16);
      f16x16 bf  = *(const f16x16*)(Wfrag + (size_t)(kt * NT + nt) * 512 + ln * 16);
      a0 = __builtin_amdgcn_wmma_f32_16x16x32_f16(false, af0, false, bf, (short)0, a0, false, false);
      a1 = __builtin_amdgcn_wmma_f32_16x16x32_f16(false, af1, false, bf, (short)0, a1, false, false);
    }
    const int col = nt * 16 + (ln & 15);
    const float s = sc[col], o = of[col];
    float m = -3.4e38f;
#pragma unroll
    for (int r = 0; r < 8; ++r) {
      float y0 = a0[r] * s + o; y0 = (y0 >= 0.f) ? y0 : SLOPE * y0;
      float y1 = a1[r] * s + o; y1 = (y1 >= 0.f) ? y1 : SLOPE * y1;
      m = fmaxf(m, fmaxf(y0, y1));
    }
    m = fmaxf(m, __shfl_xor(m, 16, 32));             // combine lane halves (wave32)
    if (ln < 16) {
      const long gp = (long)blockIdx.x * P + pp;
      if (outF) outF[gp * COUT + col] = m;
      cat[gp * 512 + catOff + col] = (_Float16)m;
    }
  }
}

// ---------------------------------------------------------------------------
// Generic WMMA GEMM + fused scale/offset (+LeakyReLU): out = act(A @ B * s + o)
// A: (M x K) f16 row-major.  Bfrag: fragment-ordered f16 weights.
// Workgroup: 64 rows x (NTB*16) cols. Templated on NTB so every wave owns a
// static tile list (no predication around WMMA); a wave's tiles share one nt,
// so the B fragment is one (CSE'd) global load per k-step.
// ---------------------------------------------------------------------------
template <int NTB>
__global__ __launch_bounds__(256) void gemm_bn_kernel(const _Float16* __restrict__ A,
                                                      const _Float16* __restrict__ Bfrag,
                                                      int Kdim, int NpadTiles,
                                                      const float* __restrict__ sc,
                                                      const float* __restrict__ of,
                                                      int doLrelu, int Nreal,
                                                      float* __restrict__ outF, long oStride, long oCol0,
                                                      _Float16* __restrict__ outH, int hBegin, int hCols) {
  constexpr int WT = (4 * NTB) / 8;                  // tiles per wave
  extern __shared__ char smemraw[];
  _Float16* sA = (_Float16*)smemraw;                 // 4 mtiles * 512 halves
  const int tid = threadIdx.x;
  const int wv = __builtin_amdgcn_readfirstlane(tid >> 5);
  const int ln = tid & 31;
  const long mBase = (long)blockIdx.x * 64;
  const int ntB0 = blockIdx.y * NTB;

  // staging map: thread -> (row r, 8-halve chunk c8)
  const int r = tid >> 2, c8 = tid & 3;
  const int dOff = (r >> 4) * 512 + ((r & 15) + 16 * (c8 >> 1)) * 16 + 8 * (c8 & 1);

  f32x8 acc[WT] = {};
  const int ktiles = Kdim >> 5;
  for (int kt = 0; kt < ktiles; ++kt) {
    __syncthreads();
    f16x8 av = *(const f16x8*)(A + (mBase + r) * (size_t)Kdim + kt * 32 + c8 * 8);
    *(f16x8*)(sA + dOff) = av;
    __syncthreads();
#pragma unroll
    for (int tt = 0; tt < WT; ++tt) {
      const int t2 = wv * WT + tt;
      const int mt = t2 & 3, nt = t2 >> 2;           // WT=4: nt==wv for all tt
      f16x16 af = *(const f16x16*)(sA + mt * 512 + ln * 16);
      f16x16 bf = *(const f16x16*)(Bfrag + (size_t)(kt * NpadTiles + ntB0 + nt) * 512 + ln * 16);
      acc[tt] = __builtin_amdgcn_wmma_f32_16x16x32_f16(false, af, false, bf, (short)0, acc[tt], false, false);
    }
  }

#pragma unroll
  for (int tt = 0; tt < WT; ++tt) {
    const int t2 = wv * WT + tt;
    const int mt = t2 & 3, nt = t2 >> 2;
    const int colG = (ntB0 + nt) * 16 + (ln & 15);
    const float s = sc[colG], o = of[colG];
    const long row0 = mBase + mt * 16 + 8 * (ln >> 4);
#pragma unroll
    for (int rr = 0; rr < 8; ++rr) {
      float y = acc[tt][rr] * s + o;
      if (doLrelu) y = (y >= 0.f) ? y : SLOPE * y;
      const long row = row0 + rr;
      if (outF && colG < Nreal) outF[row * oStride + oCol0 + colG] = y;
      if (outH && colG >= hBegin && colG < hBegin + hCols)
        outH[row * (long)hCols + (colG - hBegin)] = (_Float16)y;
    }
  }
}

// ---------------------------------------------------------------------------
extern "C" void kernel_launch(void* const* d_in, const int* in_sizes, int n_in,
                              void* d_out, int out_size, void* d_ws, size_t ws_size,
                              hipStream_t stream) {
  (void)in_sizes; (void)n_in; (void)out_size; (void)ws_size;
  const float* x    = (const float*)d_in[0];
  const float* W1   = (const float*)d_in[1];
  const float* g1   = (const float*)d_in[2];
  const float* b1   = (const float*)d_in[3];
  const float* W2   = (const float*)d_in[4];
  const float* g2   = (const float*)d_in[5];
  const float* b2   = (const float*)d_in[6];
  const float* W3   = (const float*)d_in[7];
  const float* g3   = (const float*)d_in[8];
  const float* b3   = (const float*)d_in[9];
  const float* W4   = (const float*)d_in[10];
  const float* g4   = (const float*)d_in[11];
  const float* b4   = (const float*)d_in[12];
  const float* W5   = (const float*)d_in[13];
  const float* g5   = (const float*)d_in[14];
  const float* b5   = (const float*)d_in[15];
  const float* fw   = (const float*)d_in[16];
  const float* fb   = (const float*)d_in[17];
  const float* fg   = (const float*)d_in[18];
  const float* fbt  = (const float*)d_in[19];
  const float* semw = (const float*)d_in[20];
  const float* semb = (const float*)d_in[21];
  float* out = (float*)d_out;

  const size_t PTS = (size_t)BB * NP;                // 32768
  char* p = (char*)d_ws;
  auto carve = [&](size_t bytes) { char* r = p; p += (bytes + 255) & ~(size_t)255; return r; };

  float*    xt   = (float*)carve(PTS * 16 * 4);
  float*    x1   = (float*)carve(PTS * 64 * 4);
  float*    x2   = (float*)carve(PTS * 64 * 4);
  float*    x3   = (float*)carve(PTS * 128 * 4);
  float*    x4   = (float*)carve(PTS * 256 * 4);
  int*      idx  = (int*)carve(PTS * KN * 4);
  _Float16* cat  = (_Float16*)carve(PTS * 512 * 2);
  _Float16* x5   = (_Float16*)carve(PTS * 512 * 2);
  _Float16* f3   = (_Float16*)carve(PTS * 256 * 2);
  _Float16* W1f  = (_Float16*)carve((size_t)(32/32)*(64/16)   * 512 * 2);
  _Float16* W2f  = (_Float16*)carve((size_t)(128/32)*(64/16)  * 512 * 2);
  _Float16* W3f  = (_Float16*)carve((size_t)(128/32)*(128/16) * 512 * 2);
  _Float16* W4f  = (_Float16*)carve((size_t)(256/32)*(256/16) * 512 * 2);
  _Float16* W5f  = (_Float16*)carve((size_t)(512/32)*(512/16) * 512 * 2);
  _Float16* fwf  = (_Float16*)carve((size_t)(512/32)*(1024/16)* 512 * 2);
  _Float16* semf = (_Float16*)carve((size_t)(256/32)*(32/16)  * 512 * 2);
  float* sc1 = (float*)carve(64*4);    float* of1 = (float*)carve(64*4);
  float* sc2 = (float*)carve(64*4);    float* of2 = (float*)carve(64*4);
  float* sc3 = (float*)carve(128*4);   float* of3 = (float*)carve(128*4);
  float* sc4 = (float*)carve(256*4);   float* of4 = (float*)carve(256*4);
  float* sc5 = (float*)carve(512*4);   float* of5 = (float*)carve(512*4);
  float* scH = (float*)carve(1024*4);  float* ofH = (float*)carve(1024*4);
  float* scS = (float*)carve(32*4);    float* ofS = (float*)carve(32*4);

  // ---- prep ----
  transpose_pad_kernel<<<(BB*NP*16 + 255)/256, 256, 0, stream>>>(x, xt);
  auto packN = [](int kp, int np_) { return ((kp/32)*(np_/16)*512 + 255)/256; };
  pack_b_kernel<<<packN(32, 64),   256, 0, stream>>>(W1,   W1f,  64,  18,  32,   64);
  pack_b_kernel<<<packN(128,64),   256, 0, stream>>>(W2,   W2f,  64,  128, 128,  64);
  pack_b_kernel<<<packN(128,128),  256, 0, stream>>>(W3,   W3f,  128, 128, 128,  128);
  pack_b_kernel<<<packN(256,256),  256, 0, stream>>>(W4,   W4f,  256, 256, 256,  256);
  pack_b_kernel<<<packN(512,512),  256, 0, stream>>>(W5,   W5f,  512, 512, 512,  512);
  pack_b_kernel<<<packN(512,1024), 256, 0, stream>>>(fw,   fwf,  1024,512, 512,  1024);
  pack_b_kernel<<<packN(256,32),   256, 0, stream>>>(semw, semf, 20,  256, 256,  32);
  bn_params_kernel<<<1, 256, 0, stream>>>(g1, b1, nullptr, sc1, of1, 64, 64);
  bn_params_kernel<<<1, 256, 0, stream>>>(g2, b2, nullptr, sc2, of2, 64, 64);
  bn_params_kernel<<<1, 256, 0, stream>>>(g3, b3, nullptr, sc3, of3, 128, 128);
  bn_params_kernel<<<1, 256, 0, stream>>>(g4, b4, nullptr, sc4, of4, 256, 256);
  bn_params_kernel<<<2, 256, 0, stream>>>(g5, b5, nullptr, sc5, of5, 512, 512);
  bn_params_kernel<<<4, 256, 0, stream>>>(fg, fbt, fb,     scH, ofH, 1024, 1024);
  bn_params_kernel<<<1, 256, 0, stream>>>(nullptr, nullptr, semb, scS, ofS, 32, 20);

  // ---- EdgeConv pipeline (knn -> fused gather+GEMM+BN+LReLU+max) ----
  dim3 knnG(NP/128, BB);
  knn_kernel<16><<<knnG, 128, 64*16*4, stream>>>(xt, idx);
  edgeconv_kernel<9, 16, 64><<<PTS/8, 256, 8*2*1*512*2, stream>>>(xt, idx, W1f, sc1, of1, x1, cat, 0);
  knn_kernel<64><<<knnG, 128, 64*64*4, stream>>>(x1, idx);
  edgeconv_kernel<64, 64, 64><<<PTS/8, 256, 8*2*4*512*2, stream>>>(x1, idx, W2f, sc2, of2, x2, cat, 64);
  knn_kernel<64><<<knnG, 128, 64*64*4, stream>>>(x2, idx);
  edgeconv_kernel<64, 64, 128><<<PTS/8, 256, 8*2*4*512*2, stream>>>(x2, idx, W3f, sc3, of3, x3, cat, 128);
  knn_kernel<128><<<knnG, 128, 64*128*4, stream>>>(x3, idx);
  edgeconv_kernel<128, 128, 256><<<PTS/8, 256, 8*2*8*512*2, stream>>>(x3, idx, W4f, sc4, of4, x4, cat, 256);

  // ---- conv5: x5 = lrelu(bn(cat @ W5^T)), f16 out for heads ----
  gemm_bn_kernel<8><<<dim3(PTS/64, 512/128), 256, 4*512*2, stream>>>(
      cat, W5f, 512, 512/16, sc5, of5, 1, 512,
      nullptr, 0, 0, x5, 0, 512);

  // ---- 4 heads fused as one GEMM: d_out cols 0..1023; f16 copy of head3 ----
  gemm_bn_kernel<8><<<dim3(PTS/64, 1024/128), 256, 4*512*2, stream>>>(
      x5, fwf, 512, 1024/16, scH, ofH, 1, 1024,
      out, 1044, 0, f3, 768, 256);

  // ---- semantic classifier: d_out cols 1024..1043 ----
  gemm_bn_kernel<2><<<dim3(PTS/64, 1), 256, 4*512*2, stream>>>(
      f3, semf, 256, 32/16, scS, ofS, 0, 20,
      out, 1044, 1024, nullptr, 0, 0);
}
